// GetAttentionHiddens_57114475102769
// MI455X (gfx1250) — matrix operand order repaired and checked by reading, not compiled, not run on hardware
//
#include <hip/hip_runtime.h>

#define BB 8
#define LL 2048
#define DD 256
#define HH 128

typedef _Float16 v16h __attribute__((ext_vector_type(16)));
typedef _Float16 v8h  __attribute__((ext_vector_type(8)));
typedef float    v8f  __attribute__((ext_vector_type(8)));
typedef float    v4f  __attribute__((ext_vector_type(4)));
typedef int      v4i  __attribute__((ext_vector_type(4)));

typedef __attribute__((address_space(1))) v4i* gptr4i;
typedef __attribute__((address_space(3))) v4i* lptr4i;

static __device__ inline v8f wmma16(v16h a, v16h b, v8f c) {
  // D = A(16x32 f16) * B(32x16 f16) + C(16x16 f32)
  return __builtin_amdgcn_wmma_f32_16x16x32_f16(false, a, false, b, (short)0, c,
                                                false, false);
}

static __device__ inline v16h comb(v8h lo, v8h hi) {
  return __builtin_shufflevector(lo, hi, 0, 1, 2, 3, 4, 5, 6, 7,
                                 8, 9, 10, 11, 12, 13, 14, 15);
}

static __device__ inline v8h cvt8(const float* p) {
  v4f a = *(const v4f*)p;
  v4f b = *(const v4f*)(p + 4);
  v8h r;
#pragma unroll
  for (int i = 0; i < 4; ++i) { r[i] = (_Float16)a[i]; r[4 + i] = (_Float16)b[i]; }
  return r;
}

static __device__ inline v16h cvt16(const float* p) {
  return comb(cvt8(p), cvt8(p + 8));
}

// ---- CDNA5 async global->LDS copy (ASYNCcnt path), with portable fallback ----
#if __has_builtin(__builtin_amdgcn_global_load_async_to_lds_b128)
#define HAVE_ASYNC_LDS 1
#else
#define HAVE_ASYNC_LDS 0
#endif

static __device__ inline void cp16_g2lds(_Float16* lds, const _Float16* g) {
#if HAVE_ASYNC_LDS
  __builtin_amdgcn_global_load_async_to_lds_b128((gptr4i)(void*)g,
                                                 (lptr4i)(void*)lds, 0, 0);
#else
  *(v8h*)lds = *(const v8h*)g;
#endif
}

template <int N>
static __device__ inline void wait_async() {
#if HAVE_ASYNC_LDS
#if __has_builtin(__builtin_amdgcn_s_wait_asynccnt)
  __builtin_amdgcn_s_wait_asynccnt((unsigned short)N);
#else
  if constexpr (N == 0)
    asm volatile("s_wait_asynccnt 0x0" ::: "memory");
  else
    asm volatile("s_wait_asynccnt 0xc" ::: "memory");
#endif
#endif
}

// ---------------------------------------------------------------------------
// Kernel 1: R1 = relu(x1 U^T), R2d = relu(x2 U^T) * d, both stored f16.
// One wave per 16-row tile. K = D = 256 -> 8 chained WMMAs per 16x16 tile.
// ---------------------------------------------------------------------------
__global__ __launch_bounds__(256) void proj_kernel(
    const float* __restrict__ x1, const float* __restrict__ x2,
    const float* __restrict__ U, const float* __restrict__ dvec,
    _Float16* __restrict__ R1, _Float16* __restrict__ R2d) {
  const int lane = threadIdx.x & 31;
  const int wave = threadIdx.x >> 5;
  const int tile = blockIdx.x * 8 + wave;  // 16-row tiles over [B*L1 | B*L2]
  const int nT1 = (BB * LL) / 16;
  const bool is2 = tile >= nT1;
  const float* src = is2 ? x2 : x1;
  _Float16* dst = is2 ? R2d : R1;
  const long row0 = (long)(is2 ? tile - nT1 : tile) * 16;

  const int n = lane & 15;
  const int hi = lane >> 4;
  const int ka = hi ? 16 : 0;  // A-layout K base (run of 16)
  const int kb = hi ? 16 : 0;  // B-layout K base (run of 16)

  v16h A[8];
  const float* rp = src + (row0 + n) * DD;
#pragma unroll
  for (int c = 0; c < 8; ++c)
    A[c] = comb(cvt8(rp + 32 * c + ka), cvt8(rp + 32 * c + ka + 8));

#pragma unroll
  for (int t = 0; t < HH / 16; ++t) {
    const int h = 16 * t + n;
    const float* up = U + (long)h * DD + kb;
    v8f acc;
#pragma unroll
    for (int i = 0; i < 8; ++i) acc[i] = 0.f;
#pragma unroll
    for (int c = 0; c < 8; ++c) acc = wmma16(A[c], cvt16(up + 32 * c), acc);
    const float dh = is2 ? dvec[h] : 1.0f;
#pragma unroll
    for (int r = 0; r < 8; ++r) {
      float v = acc[r];
      v = v > 0.f ? v : 0.f;
      v *= dh;
      dst[(row0 + r + 8 * hi) * HH + h] = (_Float16)v;
    }
  }
}

// ---------------------------------------------------------------------------
// Kernel 2: transpose+convert x2 [B][L2][D] f32 -> X2Tg [B][D][L2] f16, so
// the attention value tiles become contiguous 16B chunks (async-LDS friendly).
// ---------------------------------------------------------------------------
__global__ __launch_bounds__(256) void transpose_kernel(
    const float* __restrict__ x2, _Float16* __restrict__ X2Tg) {
  __shared__ float tile[32][33];
  const int b = blockIdx.z;
  const int j0 = blockIdx.x * 32;
  const int d0 = blockIdx.y * 32;
  const int tx = threadIdx.x & 31;
  const int ty = threadIdx.x >> 5;  // 0..7
#pragma unroll
  for (int k = 0; k < 4; ++k) {
    const int j = ty + 8 * k;
    tile[j][tx] = x2[(size_t)(b * LL + j0 + j) * DD + d0 + tx];
  }
  __syncthreads();
#pragma unroll
  for (int k = 0; k < 4; ++k) {
    const int d = ty + 8 * k;
    X2Tg[(size_t)(b * DD + d0 + d) * LL + j0 + tx] = (_Float16)tile[tx][d];
  }
}

// ---------------------------------------------------------------------------
// Kernel 3: flash attention. Block = 128 threads (4 waves), 64 query rows of
// one batch. 32-key chunks stream through double-buffered LDS tiles filled by
// async global->LDS copies; chunk i+1's copies overlap chunk i's compute.
// ---------------------------------------------------------------------------
#define KROWS (32 * 136)  // key buffer elems (padded stride 136)
#define VROWS (DD * 40)   // value buffer elems (padded stride 40)

__global__ __launch_bounds__(128) void attn_kernel(
    const _Float16* __restrict__ R1, const _Float16* __restrict__ R2d,
    const _Float16* __restrict__ X2Tg, const unsigned char* __restrict__ mask,
    float* __restrict__ out) {
  __shared__ _Float16 R2s[2 * KROWS];     // keys   [buf][j][h]
  __shared__ _Float16 X2T[2 * VROWS];     // values [buf][d][j]
  __shared__ _Float16 Pbuf[4 * 16 * 40];  // per-wave prob tile [m][k]

  const int tid = threadIdx.x;
  const int lane = tid & 31;
  const int wave = tid >> 5;
  const int b = blockIdx.x >> 5;          // 32 blocks per batch
  const int i0 = (blockIdx.x & 31) * 64;  // 4 waves x 16 rows

  const int n = lane & 15;
  const int hi = lane >> 4;
  const int ka = hi ? 16 : 0;
  const int kb = hi ? 16 : 0;

  const _Float16* R2base = R2d + (size_t)b * LL * HH;
  const _Float16* Vbase = X2Tg + (size_t)b * DD * LL;

  // 12 async 16B copies per thread per chunk (4 key + 8 value).
  auto stage = [&](int j0, int buf) {
    _Float16* kd = R2s + buf * KROWS;
    _Float16* vd = X2T + buf * VROWS;
#pragma unroll
    for (int it = 0; it < 4; ++it) {
      const int idx = it * 128 + tid;
      const int row = idx >> 4;   // 0..31
      const int part = idx & 15;  // 0..15
      cp16_g2lds(kd + row * 136 + part * 8,
                 R2base + (size_t)(j0 + row) * HH + part * 8);
    }
#pragma unroll
    for (int it = 0; it < 8; ++it) {
      const int idx = it * 128 + tid;
      const int d = idx >> 2;    // 0..255
      const int part = idx & 3;  // 0..3
      cp16_g2lds(vd + d * 40 + part * 8,
                 Vbase + (size_t)d * LL + j0 + part * 8);
    }
  };

  // Preload this wave's 16 query rows (16 x 128 f16) in A layout.
  v16h A1[4];
  {
    const _Float16* r1p = R1 + (size_t)(b * LL + i0 + wave * 16 + n) * HH;
#pragma unroll
    for (int c = 0; c < 4; ++c)
      A1[c] = comb(*(const v8h*)(r1p + 32 * c + ka),
                   *(const v8h*)(r1p + 32 * c + ka + 8));
  }

  v8f O[16];
#pragma unroll
  for (int t = 0; t < 16; ++t) {
#pragma unroll
    for (int r = 0; r < 8; ++r) O[t][r] = 0.f;
  }
  float mrow[8], lrow[8];
#pragma unroll
  for (int r = 0; r < 8; ++r) { mrow[r] = -3.0e38f; lrow[r] = 0.f; }

  _Float16* Pw = Pbuf + wave * 16 * 40;

  stage(0, 0);  // prologue: chunk 0 in flight

  const int NCH = LL / 32;  // 64 chunks
  for (int ic = 0; ic < NCH; ++ic) {
    const int j0 = ic * 32;
    const int cur = ic & 1;
    if (ic + 1 < NCH) {
      stage(j0 + 32, cur ^ 1);  // overlap next chunk's copies with this compute
      wait_async<12>();         // oldest 12 (chunk ic) have landed
    } else {
      wait_async<0>();
    }
    __syncthreads();  // publish chunk ic's LDS to all waves

    const _Float16* kcur = R2s + cur * KROWS;
    const _Float16* vcur = X2T + cur * VROWS;

    // ---- scores: S = A1(16x128) * keys^T(128x32), two 16-col subtiles ----
    v8f S0, S1;
#pragma unroll
    for (int r = 0; r < 8; ++r) { S0[r] = 0.f; S1[r] = 0.f; }
#pragma unroll
    for (int c = 0; c < 4; ++c) {
      const _Float16* bp = kcur + n * 136 + 32 * c + kb;
      S0 = wmma16(A1[c], comb(*(const v8h*)bp, *(const v8h*)(bp + 8)), S0);
    }
#pragma unroll
    for (int c = 0; c < 4; ++c) {
      const _Float16* bp = kcur + (16 + n) * 136 + 32 * c + kb;
      S1 = wmma16(A1[c], comb(*(const v8h*)bp, *(const v8h*)(bp + 8)), S1);
    }

    // ---- padding mask (selects, not branches: EXEC stays full) ----
    const int mbase = b * LL + j0;
    const bool mk0 = mask[mbase + n] != 0;
    const bool mk1 = mask[mbase + 16 + n] != 0;
#pragma unroll
    for (int r = 0; r < 8; ++r) {
      S0[r] = mk0 ? -1.0e30f : S0[r];
      S1[r] = mk1 ? -1.0e30f : S1[r];
    }

    // ---- online softmax stats; each row lives in one 16-lane half ----
    float sc[8];
#pragma unroll
    for (int r = 0; r < 8; ++r) {
      float t = fmaxf(S0[r], S1[r]);
#pragma unroll
      for (int d = 1; d < 16; d <<= 1) t = fmaxf(t, __shfl_xor(t, d, 32));
      const float mn = fmaxf(mrow[r], t);
      sc[r] = __expf(mrow[r] - mn);
      mrow[r] = mn;
      const float p0 = __expf(S0[r] - mn);
      const float p1 = __expf(S1[r] - mn);
      S0[r] = p0;
      S1[r] = p1;
      float s = p0 + p1;
#pragma unroll
      for (int d = 1; d < 16; d <<= 1) s += __shfl_xor(s, d, 32);
      lrow[r] = lrow[r] * sc[r] + s;
    }

    // ---- rescale running accumulators ----
#pragma unroll
    for (int t = 0; t < 16; ++t) {
#pragma unroll
      for (int r = 0; r < 8; ++r) O[t][r] *= sc[r];
    }

    // ---- C-layout -> A-layout via per-wave LDS round trip ----
#pragma unroll
    for (int r = 0; r < 8; ++r) {
      Pw[(r + 8 * hi) * 40 + n] = (_Float16)S0[r];
      Pw[(r + 8 * hi) * 40 + 16 + n] = (_Float16)S1[r];
    }
    const _Float16* pp = Pw + n * 40 + ka;
    v16h Pa = comb(*(const v8h*)pp, *(const v8h*)(pp + 8));

    // ---- O += P(16x32) * V(32x16) for each of 16 d-tiles ----
#pragma unroll
    for (int t = 0; t < 16; ++t) {
      const _Float16* xp = vcur + (16 * t + n) * 40 + kb;
      O[t] = wmma16(Pa, comb(*(const v8h*)xp, *(const v8h*)(xp + 8)), O[t]);
    }
    __syncthreads();  // don't let chunk ic+2's copies overwrite buf `cur` early
  }

  // ---- finalize: divide by softmax denominator, store f32 ----
#pragma unroll
  for (int r = 0; r < 8; ++r) lrow[r] = 1.f / lrow[r];
  const size_t obase = (size_t)(b * LL + i0 + wave * 16) * DD;
#pragma unroll
  for (int t = 0; t < 16; ++t) {
#pragma unroll
    for (int r = 0; r < 8; ++r)
      out[obase + (size_t)(r + 8 * hi) * DD + 16 * t + n] = O[t][r] * lrow[r];
  }
}

extern "C" void kernel_launch(void* const* d_in, const int* in_sizes, int n_in,
                              void* d_out, int out_size, void* d_ws,
                              size_t ws_size, hipStream_t stream) {
  const float* x1 = (const float*)d_in[0];
  const float* x2 = (const float*)d_in[1];
  const unsigned char* x2_mask = (const unsigned char*)d_in[2];
  const float* U = (const float*)d_in[3];
  const float* dv = (const float*)d_in[4];

  _Float16* R1 = (_Float16*)d_ws;               // [B*L1, H]  f16 (4 MB)
  _Float16* R2d = R1 + (size_t)BB * LL * HH;    // [B*L2, H]  f16 (4 MB)
  _Float16* X2Tg = R2d + (size_t)BB * LL * HH;  // [B, D, L2] f16 (8 MB)

  proj_kernel<<<(2 * BB * LL / 16) / 8, 256, 0, stream>>>(x1, x2, U, dv, R1,
                                                          R2d);
  transpose_kernel<<<dim3(LL / 32, DD / 32, BB), 256, 0, stream>>>(x2, X2Tg);
  attn_kernel<<<BB * (LL / 64), 128, 0, stream>>>(R1, R2d, X2Tg, x2_mask,
                                                  (float*)d_out);
  (void)in_sizes; (void)n_in; (void)out_size; (void)ws_size;
}
